// MultivariateNormalMixtureMLP_54803782697418
// MI455X (gfx1250) — matrix operand order repaired and checked by reading, not compiled, not run on hardware
//
#include <hip/hip_runtime.h>

typedef __attribute__((ext_vector_type(16))) __bf16 v16bf;
typedef __attribute__((ext_vector_type(2)))  __bf16 bf16x2;
typedef __attribute__((ext_vector_type(2)))  float  f32x2;
typedef __attribute__((ext_vector_type(8)))  float  f32x8;
typedef unsigned int uint32;

// pack two f32 -> packed bf16 (RNE) via hardware v_cvt_pk_bf16_f32 when available
__device__ __forceinline__ uint32 pk_bf16(float lo, float hi) {
#if __has_builtin(__builtin_amdgcn_cvt_pk_bf16_f32)
    bf16x2 p = __builtin_amdgcn_cvt_pk_bf16_f32(lo, hi);
#else
    f32x2 f; f.x = lo; f.y = hi;
    bf16x2 p = __builtin_convertvector(f, bf16x2);
#endif
    return __builtin_bit_cast(uint32, p);
}

#define LDS_STRIDE 40   // 32 bf16 payload + pad; rows stay 16B-aligned (80B)

// C[M,N] = act(A[M,K] @ W[K,N] + bias[N]); f32 in/out, bf16 WMMA with f32 accum.
// N, K compile-time (all our shapes: N%64==0, K%32==0); M%128==0 runtime.
template <bool RELU, int N, int K>
__global__ __launch_bounds__(256) void gemm_bias_kernel(
    const float* __restrict__ A, const float* __restrict__ W,
    const float* __restrict__ bias, float* __restrict__ C, int M)
{
    (void)M;
    __shared__ alignas(16) unsigned short lA[2][128 * LDS_STRIDE];
    __shared__ alignas(16) unsigned short lB[2][64 * LDS_STRIDE];

    const int tid  = threadIdx.x;
    const int wave = tid >> 5, lane = tid & 31;
    const int waveM = wave >> 1, waveN = wave & 1;       // 4x2 wave grid, 32x32 each
    const int bm = blockIdx.y * 128, bn = blockIdx.x * 64;

    // A loader: 2 threads/row, 16 contiguous floats each
    const int rowA = tid >> 1, halfA = tid & 1;
    // B loader: thread owns one local column nL, 8 contiguous K values (group kg)
    const int nL = tid & 63, kg = tid >> 6;

    const int hi = lane >> 4, ln = lane & 15;

    const float* gA = A + (size_t)(bm + rowA) * K + halfA * 16;
    const float* gB = W + (size_t)kg * 8 * N + bn + nL;

    f32x8  acc[2][2] = {};
    uint32 aR[8], bR[4];

    auto loadRegs = [&](int k0) {
        const float4* pa = (const float4*)(gA + k0);
#pragma unroll
        for (int q = 0; q < 4; ++q) {
            float4 f = pa[q];
            aR[q * 2]     = pk_bf16(f.x, f.y);
            aR[q * 2 + 1] = pk_bf16(f.z, f.w);
        }
        const float* pb = gB + (size_t)k0 * N;
        float fv[8];
#pragma unroll
        for (int j = 0; j < 8; ++j) fv[j] = pb[(size_t)j * N];
#pragma unroll
        for (int j = 0; j < 4; ++j) bR[j] = pk_bf16(fv[2 * j], fv[2 * j + 1]);
    };
    auto storeLds = [&](int buf) {
        uint4* da = (uint4*)&lA[buf][rowA * LDS_STRIDE + halfA * 16];
        da[0] = make_uint4(aR[0], aR[1], aR[2], aR[3]);
        da[1] = make_uint4(aR[4], aR[5], aR[6], aR[7]);
        *(uint4*)&lB[buf][nL * LDS_STRIDE + kg * 8] =
            make_uint4(bR[0], bR[1], bR[2], bR[3]);
    };

    constexpr int T = K / 32;
    loadRegs(0);
    storeLds(0);

#pragma unroll 2
    for (int t = 0; t < T; ++t) {
        __syncthreads();                       // tile t visible in buf (t&1)
        const int cur = t & 1;
        if (t + 1 < T) loadRegs((t + 1) * 32); // prefetch next tile into regs

        v16bf fa[2], fb[2];
#pragma unroll
        for (int i = 0; i < 2; ++i) {          // A frag: K = hi*8..+7, 16+hi*8..+7
            const int r = waveM * 32 + i * 16 + ln;
            uint4* q = (uint4*)&fa[i];
            q[0] = *(const uint4*)&lA[cur][r * LDS_STRIDE + hi * 8];
            q[1] = *(const uint4*)&lA[cur][r * LDS_STRIDE + 16 + hi * 8];
        }
#pragma unroll
        for (int j = 0; j < 2; ++j) {          // B frag: K = hi*8..+7, 16+hi*8..+7 (transposed LDS)
            const int n = waveN * 32 + j * 16 + ln;
            uint4* q = (uint4*)&fb[j];
            q[0] = *(const uint4*)&lB[cur][n * LDS_STRIDE + hi * 8];
            q[1] = *(const uint4*)&lB[cur][n * LDS_STRIDE + 16 + hi * 8];
        }
#pragma unroll
        for (int i = 0; i < 2; ++i)
#pragma unroll
            for (int j = 0; j < 2; ++j)
                acc[i][j] = __builtin_amdgcn_wmma_f32_16x16x32_bf16(
                    false, fa[i], false, fb[j], (short)0, acc[i][j], false, false);

        if (t + 1 < T) storeLds(cur ^ 1);      // safe: other buffer; barrier guards reuse
    }

    // epilogue: bias (+ReLU); constant N -> immediate store offsets
#pragma unroll
    for (int i = 0; i < 2; ++i) {
        const int m0 = bm + waveM * 32 + i * 16 + hi * 8;
#pragma unroll
        for (int j = 0; j < 2; ++j) {
            const int n  = bn + waveN * 32 + j * 16 + ln;
            const float bv = bias[n];
            float* Cp = C + (size_t)m0 * N + n;
#pragma unroll
            for (int r = 0; r < 8; ++r) {
                float v = acc[i][j][r] + bv;
                if (RELU) v = fmaxf(v, 0.0f);
                Cp[(size_t)r * N] = v;
            }
        }
    }
}

// pi head: logits = h2 @ Wpi + bpi (K=512, 8 outputs), fused softmax. One wave per row.
__global__ __launch_bounds__(256) void pi_softmax_kernel(
    const float* __restrict__ h, const float* __restrict__ W,
    const float* __restrict__ b, float* __restrict__ out)
{
    const int row  = blockIdx.x * 8 + (threadIdx.x >> 5);
    const int lane = threadIdx.x & 31;
    const float* hr = h + (size_t)row * 512;

    float acc[8] = {};
    for (int k = lane; k < 512; k += 32) {
        const float hv = hr[k];
        const float* wr = W + k * 8;
#pragma unroll
        for (int j = 0; j < 8; ++j) acc[j] += hv * wr[j];
    }
#pragma unroll
    for (int j = 0; j < 8; ++j) {
        float v = acc[j];
#pragma unroll
        for (int off = 16; off > 0; off >>= 1) v += __shfl_xor(v, off, 32);
        acc[j] = v + b[j];
    }
    if (lane == 0) {
        float m = acc[0];
#pragma unroll
        for (int j = 1; j < 8; ++j) m = fmaxf(m, acc[j]);
        float e[8], s = 0.0f;
#pragma unroll
        for (int j = 0; j < 8; ++j) { e[j] = __expf(acc[j] - m); s += e[j]; }
        const float inv = 1.0f / s;
#pragma unroll
        for (int j = 0; j < 8; ++j) out[(size_t)row * 8 + j] = e[j] * inv;
    }
}

// tril scatter: tc[b,k,136] -> out[b,k,16,16]; exp on diagonal, zero above.
// Vectorized: one float4 (4 cols of one row) per thread.
__global__ __launch_bounds__(256) void tril_kernel(
    const float* __restrict__ tc, float* __restrict__ out)
{
    const size_t q   = (size_t)blockIdx.x * 256 + threadIdx.x;
    const size_t idx = q * 4;
    const int j0 = (int)(idx & 15);
    const int i  = (int)((idx >> 4) & 15);
    const size_t bk = idx >> 8;                       // b*K + k
    const float* src = tc + bk * 136 + (i * (i + 1)) / 2;   // row-major tril row i

    float4 v;
    float* pv = &v.x;
#pragma unroll
    for (int t = 0; t < 4; ++t) {
        const int j = j0 + t;
        float x = 0.0f;
        if (j <= i) {
            x = src[j];
            if (j == i) x = __expf(x);
        }
        pv[t] = x;
    }
    *(float4*)(out + idx) = v;
}

extern "C" void kernel_launch(void* const* d_in, const int* in_sizes, int n_in,
                              void* d_out, int out_size, void* d_ws, size_t ws_size,
                              hipStream_t stream) {
    (void)in_sizes; (void)n_in; (void)out_size; (void)ws_size;

    const float* x   = (const float*)d_in[0];
    const float* W0  = (const float*)d_in[1];  const float* b0  = (const float*)d_in[2];
    const float* W1  = (const float*)d_in[3];  const float* b1  = (const float*)d_in[4];
    const float* W2  = (const float*)d_in[5];  const float* b2  = (const float*)d_in[6];
    const float* Wpi = (const float*)d_in[7];  const float* bpi = (const float*)d_in[8];
    const float* Wmu = (const float*)d_in[9];  const float* bmu = (const float*)d_in[10];
    const float* Wt  = (const float*)d_in[11]; const float* bt  = (const float*)d_in[12];

    float* out = (float*)d_out;
    float* ws  = (float*)d_ws;

    const int M = 16384;
    // workspace (floats), liveness-based reuse; peak 33,554,432 floats = 128 MiB
    float* h0 = ws;                       // M x 1024
    float* h1 = ws + 16777216;            // M x 1024
    float* h2 = ws;                       // M x 512  (reuses dead h0)
    float* tc = ws + 8388608;             // M x 1088 (reuses dead h1 region)

    float* out_pi   = out;                          // M*8
    float* out_mu   = out + (size_t)M * 8;          // M*128
    float* out_tril = out + (size_t)M * (8 + 128);  // M*2048

    const dim3 blk(256);

    // trunk
    gemm_bias_kernel<true, 1024,  256><<<dim3(16, M / 128), blk, 0, stream>>>(x,  W0, b0, h0, M);
    gemm_bias_kernel<true, 1024, 1024><<<dim3(16, M / 128), blk, 0, stream>>>(h0, W1, b1, h1, M);
    gemm_bias_kernel<true,  512, 1024><<<dim3( 8, M / 128), blk, 0, stream>>>(h1, W2, b2, h2, M);
    // heads
    pi_softmax_kernel<<<M / 8, blk, 0, stream>>>(h2, Wpi, bpi, out_pi);
    gemm_bias_kernel<false, 128, 512><<<dim3( 2, M / 128), blk, 0, stream>>>(h2, Wmu, bmu, out_mu, M);
    gemm_bias_kernel<false, 1088, 512><<<dim3(17, M / 128), blk, 0, stream>>>(h2, Wt, bt, tc, M);
    // tril epilogue: one float4 per thread (M*8*256 elems / 4)
    tril_kernel<<<(M * 8 * 256) / 1024, blk, 0, stream>>>(tc, out_tril);
}
